// Encoder_50087908606215
// MI455X (gfx1250) — matrix operand order repaired and checked by reading, not compiled, not run on hardware
//
#include <hip/hip_runtime.h>
#include <hip/hip_bf16.h>
#include <math.h>

// ---------------------------------------------------------------------------
// Problem constants (match reference)
// ---------------------------------------------------------------------------
#define BB 2
#define TT 2048
#define CC 1024
#define HH 16
#define DD 64

typedef __attribute__((ext_vector_type(16))) __bf16 bf16x16;
typedef __attribute__((ext_vector_type(8)))  float  floatx8;

union BFrag {
    bf16x16      v;
    unsigned int u32[8];
    unsigned short u16[16];
};

__device__ __forceinline__ unsigned short f2bf(float f) {
    unsigned int u = __float_as_uint(f);
    u += 0x7FFFu + ((u >> 16) & 1u);   // round-to-nearest-even
    return (unsigned short)(u >> 16);
}

// ---------------------------------------------------------------------------
// Async global -> LDS copy of 16 bytes (CDNA5 GLOBAL_LOAD_ASYNC_TO_LDS_B128,
// tracked by ASYNCcnt).
// ---------------------------------------------------------------------------
typedef int v4i_vec __attribute__((vector_size(16)));
typedef __attribute__((address_space(1))) v4i_vec* as1_v4i_t;
typedef __attribute__((address_space(3))) v4i_vec* as3_v4i_t;

__device__ __forceinline__ void async_copy16(unsigned short* lds_dst,
                                             const unsigned short* gsrc) {
#if defined(__has_builtin) && __has_builtin(__builtin_amdgcn_global_load_async_to_lds_b128)
    __builtin_amdgcn_global_load_async_to_lds_b128(
        (as1_v4i_t)(uintptr_t)gsrc,
        (as3_v4i_t)(uintptr_t)lds_dst, 0, 0);
#else
    unsigned lds_off = (unsigned)(uintptr_t)lds_dst;      // addr[31:0] = LDS offset
    unsigned long long ga = (unsigned long long)(uintptr_t)gsrc;
    asm volatile("global_load_async_to_lds_b128 %0, %1, off"
                 :: "v"(lds_off), "v"(ga) : "memory");
#endif
}

__device__ __forceinline__ void wait_async0() {
    asm volatile("s_wait_asynccnt 0" ::: "memory");
}
__device__ __forceinline__ void wait_async4() {   // allow next tile's 4 DMAs in flight
    asm volatile("s_wait_asynccnt 4" ::: "memory");
}

// ---------------------------------------------------------------------------
// Weight convert + transpose: f32 [K][N] -> bf16 [N][K]
// ---------------------------------------------------------------------------
__global__ void conv_transpose_kernel(const float* __restrict__ in,
                                      unsigned short* __restrict__ out,
                                      int K, int N) {
    size_t idx = (size_t)blockIdx.x * blockDim.x + threadIdx.x;
    size_t total = (size_t)K * N;
    if (idx >= total) return;
    int n = (int)(idx / K);
    int k = (int)(idx % K);
    out[(size_t)n * K + k] = f2bf(in[(size_t)k * N + n]);
}

// ---------------------------------------------------------------------------
// LayerNorm over C=1024, output bf16.  One 256-thread block per row.
// ---------------------------------------------------------------------------
__global__ __launch_bounds__(256) void ln_kernel(const float* __restrict__ x,
                                                 const float* __restrict__ w,
                                                 const float* __restrict__ b,
                                                 unsigned short* __restrict__ out) {
    const int row = blockIdx.x;
    const int tid = threadIdx.x;
    const float* xr = x + (size_t)row * CC;
    __shared__ float red[256];

    float v[4];
    float s = 0.f;
#pragma unroll
    for (int i = 0; i < 4; ++i) { v[i] = xr[tid + i * 256]; s += v[i]; }
    red[tid] = s; __syncthreads();
    for (int o = 128; o > 0; o >>= 1) { if (tid < o) red[tid] += red[tid + o]; __syncthreads(); }
    const float mu = red[0] * (1.0f / CC);
    __syncthreads();

    float vs = 0.f;
#pragma unroll
    for (int i = 0; i < 4; ++i) { float d = v[i] - mu; vs += d * d; }
    red[tid] = vs; __syncthreads();
    for (int o = 128; o > 0; o >>= 1) { if (tid < o) red[tid] += red[tid + o]; __syncthreads(); }
    const float rstd = rsqrtf(red[0] * (1.0f / CC) + 1e-5f);

#pragma unroll
    for (int i = 0; i < 4; ++i) {
        int c = tid + i * 256;
        out[(size_t)row * CC + c] = f2bf((v[i] - mu) * rstd * w[c] + b[c]);
    }
}

// ---------------------------------------------------------------------------
// Tiled WMMA GEMM:  C[M][N] = A[M][K](bf16) * Bt[N][K](bf16)^T  (f32 accum)
// Block tile 128(M) x 128(N), BK=32, 8 waves; wave w owns rows w*16..w*16+15
// and all 128 columns (8 accumulators).  Double-buffered LDS with one-tile-
// ahead async-to-LDS DMA (ASYNCcnt in-order completion => s_wait_asynccnt 4
// drains the current tile while the next tile stays in flight).
// ---------------------------------------------------------------------------
#define EP_STORE_BF16     0
#define EP_RES_F32        1
#define EP_BIAS_GELU_BF16 2
#define EP_BIAS_RES_F32   3

__global__ __launch_bounds__(256) void gemm_bf16_wmma(
    const unsigned short* __restrict__ A,   // [M][K] bf16
    const unsigned short* __restrict__ Bt,  // [N][K] bf16 (transposed weight)
    int M, int N, int K,
    const float* __restrict__ bias,
    const float* __restrict__ res,
    unsigned short* __restrict__ outb,
    float* __restrict__ outf,
    int mode)
{
    __shared__ unsigned short As[2][128 * 32];   // 2 x 8 KB
    __shared__ unsigned short Bs[2][128 * 32];   // 2 x 8 KB

    const int tid  = threadIdx.x;
    const int lane = tid & 31;
    const int wv   = tid >> 5;       // 0..7
    const int half = lane >> 4;      // lane-half selects K sub-range / M+8
    const int l15  = lane & 15;
    const int mBase = blockIdx.y * 128;
    const int nBase = blockIdx.x * 128;

    // per-thread staging coordinates: 2 chunks for A, 2 for B
    const int r0 = tid >> 1;               // unused; stage() uses row = idx >> 2
    const int c0 = (tid & 1) * 16;         // not used directly; see below

    floatx8 acc[8] = {};

    auto stage = [&](int buf, int kk) {
#pragma unroll
        for (int c = 0; c < 2; ++c) {
            int idx = tid + c * 256;                  // 0..511
            int r   = idx >> 2;
            int k8  = (idx & 3) * 8;
            async_copy16(&As[buf][r * 32 + k8], &A[(size_t)(mBase + r) * K + kk + k8]);
            async_copy16(&Bs[buf][r * 32 + k8], &Bt[(size_t)(nBase + r) * K + kk + k8]);
        }
    };

    const int nt = K / 32;
    stage(0, 0);                     // prologue DMA for tile 0

    for (int t = 0; t < nt; ++t) {
        const int cur = t & 1;
        __syncthreads();             // all waves done reading buf(1-cur) (tile t-1)
        if (t + 1 < nt) {
            stage(1 - cur, (t + 1) * 32);          // overlap next tile's DMA
            if (t + 2 < nt) {                      // L2 prefetch two tiles ahead
                int kk2 = (t + 2) * 32;
                __builtin_prefetch(&A[(size_t)(mBase + (tid >> 1)) * K + kk2], 0, 1);
                __builtin_prefetch(&Bt[(size_t)(nBase + (tid >> 1)) * K + kk2], 0, 1);
            }
            wait_async4();           // tile t's 4 DMAs done; tile t+1 in flight
        } else {
            wait_async0();
        }
        __syncthreads();             // all waves' tile-t DMA landed

        // A fragment: lane = M row, half selects K 0-15 / 16-31, VGPR i -> K=2i
        BFrag af;
#pragma unroll
        for (int i = 0; i < 8; ++i) {
            int kp = 16 * half + 2 * i;
            af.u32[i] = *reinterpret_cast<const unsigned int*>(
                &As[cur][(wv * 16 + l15) * 32 + kp]);
        }
#pragma unroll
        for (int j = 0; j < 8; ++j) {
            // B fragment: lane = N col, half selects K 0-15 / 16-31, VGPR i -> K=2i
            BFrag bf;
#pragma unroll
            for (int i = 0; i < 8; ++i)
                bf.u32[i] = *reinterpret_cast<const unsigned int*>(
                    &Bs[cur][(j * 16 + l15) * 32 + 16 * half + 2 * i]);
            acc[j] = __builtin_amdgcn_wmma_f32_16x16x32_bf16(
                false, af.v, false, bf.v, (short)0, acc[j], false, false);
        }
    }

    // --- epilogue: C/D layout -> row = i + 8*half, col = lane&15 ---
#pragma unroll
    for (int j = 0; j < 8; ++j) {
#pragma unroll
        for (int i = 0; i < 8; ++i) {
            int m = mBase + wv * 16 + i + 8 * half;
            int n = nBase + j * 16 + l15;
            float v = acc[j][i];
            size_t o = (size_t)m * N + n;
            if (mode == EP_STORE_BF16) {
                outb[o] = f2bf(v);
            } else if (mode == EP_RES_F32) {
                outf[o] = v + res[o];
            } else if (mode == EP_BIAS_GELU_BF16) {
                float t2 = v + bias[n];
                outb[o] = f2bf(0.5f * t2 * (1.0f + erff(t2 * 0.70710678118654752f)));
            } else { // EP_BIAS_RES_F32
                outf[o] = v + bias[n] + res[o];
            }
        }
    }
}

// ---------------------------------------------------------------------------
// Flash attention: one wave per (b, h, 16-row Q tile).
// S = (Q Kt) * 1/sqrt(D) via WMMA; streaming softmax; O += P V via WMMA.
// ---------------------------------------------------------------------------
__global__ __launch_bounds__(32) void attn_wmma(
    const unsigned short* __restrict__ qkv,  // [B*T][3C] bf16
    const int* __restrict__ mask,            // [B*T]
    unsigned short* __restrict__ y)          // [B*T][C] bf16
{
    const int qt   = blockIdx.x;   // T/16 tiles
    const int h    = blockIdx.y;
    const int b    = blockIdx.z;
    const int lane = threadIdx.x;
    const int half = lane >> 4;
    const int l15  = lane & 15;
    const float scale = 0.125f;    // 1/sqrt(64)
    const int C3 = 3 * CC;

    __shared__ unsigned short Ps[16 * 32];   // P tile bounce buffer

    const size_t base = (size_t)b * TT * C3;
    const int qoff = h * DD;
    const int koff = CC + h * DD;
    const int voff = 2 * CC + h * DD;

    // Q A-fragments (d = 0..31 and 32..63)
    BFrag qa[2];
    {
        const unsigned short* qp = qkv + base + (size_t)(qt * 16 + l15) * C3 + qoff;
#pragma unroll
        for (int f = 0; f < 2; ++f)
#pragma unroll
            for (int i = 0; i < 8; ++i) {
                int kp = 16 * half + 2 * i + f * 32;
                qa[f].u32[i] = *reinterpret_cast<const unsigned int*>(qp + kp);
            }
    }

    float m_run[8], l_run[8];
#pragma unroll
    for (int i = 0; i < 8; ++i) { m_run[i] = -INFINITY; l_run[i] = 0.f; }
    floatx8 oacc[4] = {};

    for (int kb = 0; kb < TT; kb += 32) {
        // ---- S tiles for 32 keys (two 16x16 tiles) ----
        floatx8 s[2] = {};
#pragma unroll
        for (int jt = 0; jt < 2; ++jt) {
            const unsigned short* kp_ =
                qkv + base + (size_t)(kb + jt * 16 + l15) * C3 + koff;
#pragma unroll
            for (int f = 0; f < 2; ++f) {
                BFrag kf;   // B = K^T: col=key, K dim = d (contiguous pairs)
#pragma unroll
                for (int i = 0; i < 8; ++i)
                    kf.u32[i] = *reinterpret_cast<const unsigned int*>(
                        kp_ + f * 32 + 16 * half + 2 * i);
                s[jt] = __builtin_amdgcn_wmma_f32_16x16x32_bf16(
                    false, qa[f].v, false, kf.v, (short)0, s[jt], false, false);
            }
        }
        // ---- scale + src_mask (column = key per lane) ----
        int mk0 = mask[b * TT + kb + l15];
        int mk1 = mask[b * TT + kb + 16 + l15];
#pragma unroll
        for (int i = 0; i < 8; ++i) {
            s[0][i] = mk0 ? s[0][i] * scale : -INFINITY;
            s[1][i] = mk1 ? s[1][i] * scale : -INFINITY;
        }
        // ---- online softmax (half-wave butterfly reductions per row) ----
        float pn0[8], pn1[8];
#pragma unroll
        for (int i = 0; i < 8; ++i) {
            float mx = fmaxf(s[0][i], s[1][i]);
#pragma unroll
            for (int off = 8; off > 0; off >>= 1) mx = fmaxf(mx, __shfl_xor(mx, off, 32));
            float mnew  = fmaxf(m_run[i], mx);
            float alpha = __expf(m_run[i] - mnew);
            float p0 = __expf(s[0][i] - mnew);
            float p1 = __expf(s[1][i] - mnew);
            float rs = p0 + p1;
#pragma unroll
            for (int off = 8; off > 0; off >>= 1) rs += __shfl_xor(rs, off, 32);
            l_run[i] = l_run[i] * alpha + rs;
            m_run[i] = mnew;
#pragma unroll
            for (int j = 0; j < 4; ++j) oacc[j][i] *= alpha;
            pn0[i] = p0; pn1[i] = p1;
        }
        // ---- C-layout P -> A-layout P through LDS (same-wave DS is in-order) ----
#pragma unroll
        for (int i = 0; i < 8; ++i) {
            int r = i + 8 * half;
            Ps[r * 32 + l15]      = f2bf(pn0[i]);
            Ps[r * 32 + 16 + l15] = f2bf(pn1[i]);
        }
        asm volatile("s_wait_dscnt 0" ::: "memory");
        BFrag pa;
#pragma unroll
        for (int i = 0; i < 8; ++i) {
            int kp = 16 * half + 2 * i;
            pa.u32[i] = *reinterpret_cast<const unsigned int*>(&Ps[l15 * 32 + kp]);
        }
        // ---- O += P V ----
#pragma unroll
        for (int j = 0; j < 4; ++j) {
            BFrag vf;   // B[k][n]: k = key index (row stride 3C), n = d column
#pragma unroll
            for (int i = 0; i < 8; ++i) {
                int key0 = kb + 16 * half + 2 * i;
                vf.u16[2 * i] =
                    qkv[base + (size_t)key0 * C3 + voff + j * 16 + l15];
                vf.u16[2 * i + 1] =
                    qkv[base + (size_t)(key0 + 1) * C3 + voff + j * 16 + l15];
            }
            oacc[j] = __builtin_amdgcn_wmma_f32_16x16x32_bf16(
                false, pa.v, false, vf.v, (short)0, oacc[j], false, false);
        }
    }

    // ---- normalize and store O (bf16) ----
#pragma unroll
    for (int j = 0; j < 4; ++j)
#pragma unroll
        for (int i = 0; i < 8; ++i) {
            int row = qt * 16 + i + 8 * half;
            int col = h * DD + j * 16 + l15;
            y[((size_t)b * TT + row) * CC + col] = f2bf(oacc[j][i] / l_run[i]);
        }
}

// ---------------------------------------------------------------------------
// Host orchestration
// ---------------------------------------------------------------------------
extern "C" void kernel_launch(void* const* d_in, const int* in_sizes, int n_in,
                              void* d_out, int out_size, void* d_ws, size_t ws_size,
                              hipStream_t stream) {
    (void)in_sizes; (void)n_in; (void)out_size; (void)ws_size;
    const int M = BB * TT;                    // 4096 rows

    const float* x    = (const float*)d_in[0];
    const int*   msk  = (const int*)d_in[1];
    const float* ln1w = (const float*)d_in[2];
    const float* ln1b = (const float*)d_in[3];
    const float* wat  = (const float*)d_in[4];
    const float* wpr  = (const float*)d_in[5];
    const float* ln2w = (const float*)d_in[6];
    const float* ln2b = (const float*)d_in[7];
    const float* wfc  = (const float*)d_in[8];
    const float* bfc  = (const float*)d_in[9];
    const float* wout = (const float*)d_in[10];
    const float* bout = (const float*)d_in[11];
    float* out = (float*)d_out;

    // workspace carve-up (buffers reused across pipeline stages)
    char* ws = (char*)d_ws;
    size_t off = 0;
    auto alloc = [&](size_t bytes) -> void* {
        void* p = (void*)(ws + off);
        off += (bytes + 255) & ~(size_t)255;
        return p;
    };
    unsigned short* wT   = (unsigned short*)alloc((size_t)4 * CC * CC * 2); // max weight (8.4MB)
    unsigned short* actA = (unsigned short*)alloc((size_t)M * CC * 2);      // h1 / y / h2in
    unsigned short* actB = (unsigned short*)alloc((size_t)M * 4 * CC * 2);  // qkv / h2
    float*          x1   = (float*)alloc((size_t)M * CC * 4);               // attn residual out

    const int THR = 256;
    dim3 blk(THR);

    // ---- stage 1: qkv = LN1(x) @ w_attn ----
    {
        size_t tot = (size_t)CC * 3 * CC;
        conv_transpose_kernel<<<dim3((unsigned)((tot + THR - 1) / THR)), blk, 0, stream>>>(
            wat, wT, CC, 3 * CC);
        ln_kernel<<<dim3(M), blk, 0, stream>>>(x, ln1w, ln1b, actA);
        dim3 g(3 * CC / 128, M / 128);
        gemm_bf16_wmma<<<g, blk, 0, stream>>>(actA, wT, M, 3 * CC, CC,
                                              nullptr, nullptr, actB, nullptr, EP_STORE_BF16);
    }
    // ---- stage 2: y = attention(qkv) ----
    attn_wmma<<<dim3(TT / 16, HH, BB), dim3(32), 0, stream>>>(actB, msk, actA);

    // ---- stage 3: x1 = x + y @ w_proj ----
    {
        size_t tot = (size_t)CC * CC;
        conv_transpose_kernel<<<dim3((unsigned)((tot + THR - 1) / THR)), blk, 0, stream>>>(
            wpr, wT, CC, CC);
        dim3 g(CC / 128, M / 128);
        gemm_bf16_wmma<<<g, blk, 0, stream>>>(actA, wT, M, CC, CC,
                                              nullptr, x, nullptr, x1, EP_RES_F32);
    }
    // ---- stage 4: h2 = gelu(LN2(x1) @ w_fc + b_fc) ----
    {
        size_t tot = (size_t)CC * 4 * CC;
        conv_transpose_kernel<<<dim3((unsigned)((tot + THR - 1) / THR)), blk, 0, stream>>>(
            wfc, wT, CC, 4 * CC);
        ln_kernel<<<dim3(M), blk, 0, stream>>>(x1, ln2w, ln2b, actA);
        dim3 g(4 * CC / 128, M / 128);
        gemm_bf16_wmma<<<g, blk, 0, stream>>>(actA, wT, M, 4 * CC, CC,
                                              bfc, nullptr, actB, nullptr, EP_BIAS_GELU_BF16);
    }
    // ---- stage 5: out = x1 + h2 @ w_out + b_out ----
    {
        size_t tot = (size_t)4 * CC * CC;
        conv_transpose_kernel<<<dim3((unsigned)((tot + THR - 1) / THR)), blk, 0, stream>>>(
            wout, wT, 4 * CC, CC);
        dim3 g(CC / 128, M / 128);
        gemm_bf16_wmma<<<g, blk, 0, stream>>>(actB, wT, M, CC, 4 * CC,
                                              bout, x1, nullptr, out, EP_BIAS_RES_F32);
    }
}
